// YvCrossAttentionBlock_6330781794498
// MI455X (gfx1250) — matrix operand order repaired and checked by reading, not compiled, not run on hardware
//
#include <hip/hip_runtime.h>
#include <hip/hip_bf16.h>

typedef __bf16 bf16_t;
typedef __attribute__((ext_vector_type(16))) __bf16 v16bf;
typedef __attribute__((ext_vector_type(8)))  __bf16 v8bf;
typedef __attribute__((ext_vector_type(8)))  float  v8f;

#define DEVI __device__ __forceinline__

// ---------------------------------------------------------------- fragments
DEVI v8f zero8f() {
  v8f z = {0.f, 0.f, 0.f, 0.f, 0.f, 0.f, 0.f, 0.f};
  return z;
}

DEVI v8f wmma_bf16(v16bf a, v16bf b, v8f c) {
  // emits v_wmma_f32_16x16x32_bf16
  return __builtin_amdgcn_wmma_f32_16x16x32_bf16(false, a, false, b, (short)0, c,
                                                 false, false);
}

// A-matrix 16x32 bf16 fragment: lane holds row m=lane&15.
// elements j -> K = koff + j (16 contiguous K values)
// where koff = kb + 16*(lane>=16).
DEVI v16bf load_a_frag(const bf16_t* row, int kb, int lane) {
  int koff = kb + ((lane & 16) ? 16 : 0);
  v8bf c0 = *(const v8bf*)(row + koff);
  v8bf c1 = *(const v8bf*)(row + koff + 8);
  v16bf r;
#pragma unroll
  for (int i = 0; i < 8; ++i) { r[i] = c0[i]; r[i + 8] = c1[i]; }
  return r;
}

// B-matrix 32x16 bf16 fragment: lane holds col n=lane&15.
// elements j -> K = kb + j + 16*(lane>=16)  (16 contiguous K values).
DEVI v16bf load_b_frag(const bf16_t* row, int kb, int lane) {
  int koff = kb + ((lane & 16) ? 16 : 0);
  v8bf c0 = *(const v8bf*)(row + koff);
  v8bf c1 = *(const v8bf*)(row + koff + 8);
  v16bf r;
#pragma unroll
  for (int i = 0; i < 8; ++i) { r[i] = c0[i]; r[i + 8] = c1[i]; }
  return r;
}

// ---------------------------------------------------------------- GEMM
// C[M,N] = A[M,K](bf16, row-major) * Bt[N,K](bf16, row-major = B transposed)
// Block 256 thr = 8 waves (4 M x 2 N); block tile 128x128; wave tile 32x64.
// 8 WMMA per 12 b128 loads per k-step of 32.
// Epilogue options: per-row scale, residual add, accumulate, SwiGLU fuse
// (v = silu(glu_src[idx]) * v), fp32 or bf16 output.
template <bool ACC, bool HAS_ADD, bool HAS_SCALE, bool OUT_BF16, bool GLU>
__global__ __launch_bounds__(256) void gemm_wmma_kernel(
    const bf16_t* __restrict__ A, const bf16_t* __restrict__ Bt,
    void* __restrict__ Cout, const float* __restrict__ add_src,
    const float* __restrict__ glu_src, const float* __restrict__ row_scale,
    int scale_stride, int K, int lda, int ldb, int ldc) {
  int tid = threadIdx.x;
  int lane = tid & 31;
  int w = tid >> 5;
  int wm = w & 3;    // 0..3
  int wn = w >> 2;   // 0..1
  int m0 = blockIdx.y * 128 + wm * 32;
  int n0 = blockIdx.x * 128 + wn * 64;
  int nn = lane & 15;

  v8f cf[2][4];
#pragma unroll
  for (int mt = 0; mt < 2; ++mt)
#pragma unroll
    for (int nt = 0; nt < 4; ++nt) cf[mt][nt] = zero8f();

  const bf16_t* arow[2];
  arow[0] = A + (size_t)(m0 + nn) * lda;
  arow[1] = arow[0] + (size_t)16 * lda;
  const bf16_t* brow[4];
#pragma unroll
  for (int nt = 0; nt < 4; ++nt)
    brow[nt] = Bt + (size_t)(n0 + nt * 16 + nn) * ldb;

  for (int kb = 0; kb < K; kb += 32) {
    v16bf af[2], bfr[4];
#pragma unroll
    for (int mt = 0; mt < 2; ++mt) af[mt] = load_a_frag(arow[mt], kb, lane);
#pragma unroll
    for (int nt = 0; nt < 4; ++nt) bfr[nt] = load_b_frag(brow[nt], kb, lane);
#pragma unroll
    for (int mt = 0; mt < 2; ++mt)
#pragma unroll
      for (int nt = 0; nt < 4; ++nt)
        cf[mt][nt] = wmma_bf16(af[mt], bfr[nt], cf[mt][nt]);
  }

  int half = (lane >> 4) & 1;
#pragma unroll
  for (int mt = 0; mt < 2; ++mt)
#pragma unroll
    for (int nt = 0; nt < 4; ++nt) {
#pragma unroll
      for (int r = 0; r < 8; ++r) {
        int m = m0 + mt * 16 + r + 8 * half;
        int n = n0 + nt * 16 + nn;
        size_t idx = (size_t)m * ldc + n;
        float v = cf[mt][nt][r];
        if (HAS_SCALE) v *= row_scale[(size_t)m * scale_stride];
        if (GLU) {
          float gv = glu_src[idx];
          v *= gv / (1.f + __expf(-gv));
        }
        if (HAS_ADD) v += add_src[idx];
        if (OUT_BF16) {
          ((bf16_t*)Cout)[idx] = (bf16_t)v;
        } else {
          float* C = (float*)Cout;
          if (ACC) v += C[idx];
          C[idx] = v;
        }
      }
    }
}

// ---------------------------------------------------------------- attention
// Fused flash attention. Q [B*S, 1024] bf16, K [B*L, 256] bf16,
// Vt [B,4,64,L] bf16, O [B*S, 1024] bf16. Block 128 thr = 4 waves; each wave
// owns 16 q rows (online softmax fully wave-local), block covers 64 q rows.
__global__ __launch_bounds__(128) void flash_attn_kernel(
    const bf16_t* __restrict__ Q, const bf16_t* __restrict__ Kb,
    const bf16_t* __restrict__ Vt, bf16_t* __restrict__ O, int Sq, int Lk,
    int causal) {
  __shared__ __align__(64) bf16_t psh[4][16][32];
  int tid = threadIdx.x;
  int lane = tid & 31;
  int w = tid >> 5;
  int qt = blockIdx.x;
  int h = blockIdx.y;
  int b = blockIdx.z;
  int kvh = h >> 2;
  int qbase = qt * 64 + w * 16;
  int half = (lane >> 4) & 1;
  int nn = lane & 15;

  const bf16_t* qrow = Q + (size_t)(b * Sq + qbase + nn) * 1024 + h * 64;
  v16bf qf0 = load_a_frag(qrow, 0, lane);
  v16bf qf1 = load_a_frag(qrow, 32, lane);

  float mrow[8], lrow[8];
  v8f oacc[4];
#pragma unroll
  for (int r = 0; r < 8; ++r) { mrow[r] = -3.0e38f; lrow[r] = 0.f; }
#pragma unroll
  for (int j = 0; j < 4; ++j) oacc[j] = zero8f();

  int kend = causal ? (qt * 64 + 64) : Lk;
  for (int kt = 0; kt < kend; kt += 32) {
    v8f s[2];
#pragma unroll
    for (int g = 0; g < 2; ++g) {
      const bf16_t* krow =
          Kb + (size_t)(b * Lk + kt + g * 16 + nn) * 256 + kvh * 64;
      v16bf kf0 = load_b_frag(krow, 0, lane);
      v16bf kf1 = load_b_frag(krow, 32, lane);
      v8f sc = zero8f();
      sc = wmma_bf16(qf0, kf0, sc);
      sc = wmma_bf16(qf1, kf1, sc);
      s[g] = sc;
    }
#pragma unroll
    for (int r = 0; r < 8; ++r) {
      int m = qbase + r + 8 * half;
      float v0 = s[0][r] * 0.125f;  // HD^-0.5 = 1/8
      float v1 = s[1][r] * 0.125f;
      if (causal) {
        if (kt + nn > m) v0 = -1e9f;
        if (kt + 16 + nn > m) v1 = -1e9f;
      }
      float mx = fmaxf(v0, v1);
#pragma unroll
      for (int sh = 8; sh >= 1; sh >>= 1) mx = fmaxf(mx, __shfl_xor(mx, sh, 32));
      float mnew = fmaxf(mrow[r], mx);
      float corr = __expf(mrow[r] - mnew);
      float p0 = __expf(v0 - mnew);
      float p1 = __expf(v1 - mnew);
      float rs = p0 + p1;
#pragma unroll
      for (int sh = 8; sh >= 1; sh >>= 1) rs += __shfl_xor(rs, sh, 32);
      lrow[r] = lrow[r] * corr + rs;
      mrow[r] = mnew;
#pragma unroll
      for (int j = 0; j < 4; ++j) oacc[j][r] *= corr;
      psh[w][r + 8 * half][nn] = (bf16_t)p0;
      psh[w][r + 8 * half][16 + nn] = (bf16_t)p1;
    }
    __syncthreads();
    v16bf pf = load_a_frag(&psh[w][nn][0], 0, lane);
#pragma unroll
    for (int j = 0; j < 4; ++j) {
      const bf16_t* vrow =
          Vt + ((size_t)(b * 4 + kvh) * 64 + j * 16 + nn) * (size_t)Lk + kt;
      v16bf vf = load_b_frag(vrow, 0, lane);
      oacc[j] = wmma_bf16(pf, vf, oacc[j]);
    }
    __syncthreads();
  }
#pragma unroll
  for (int j = 0; j < 4; ++j)
#pragma unroll
    for (int r = 0; r < 8; ++r) {
      int m = qbase + r + 8 * half;
      float v = oacc[j][r] / lrow[r];
      O[(size_t)(b * Sq + m) * 1024 + h * 64 + j * 16 + nn] = (bf16_t)v;
    }
}

// ---------------------------------------------------------------- elementwise
__global__ void convert_w_t_kernel(const float* __restrict__ W,
                                   bf16_t* __restrict__ Wt, int Kd, int Nd,
                                   long total) {
  long idx = (long)blockIdx.x * 256 + threadIdx.x;
  if (idx >= total) return;
  long per = (long)Kd * Nd;
  long mat = idx / per;
  long rem = idx - mat * per;
  int k = (int)(rem / Nd);
  int n = (int)(rem - (long)k * Nd);
  Wt[mat * per + (long)n * Kd + k] = (bf16_t)W[idx];
}

__global__ void cast_bf16_kernel(const float* __restrict__ X,
                                 bf16_t* __restrict__ Y, long n) {
  long idx = (long)blockIdx.x * 256 + threadIdx.x;
  if (idx >= n) return;
  Y[idx] = (bf16_t)X[idx];
}

__global__ void transpose_v_kernel(const bf16_t* __restrict__ V,
                                   bf16_t* __restrict__ Vt, int Lk,
                                   long total) {
  long idx = (long)blockIdx.x * 256 + threadIdx.x;
  if (idx >= total) return;
  int hd = (int)(idx & 63);
  int kvh = (int)((idx >> 6) & 3);
  long bl = idx >> 8;
  int l = (int)(bl % Lk);
  int b = (int)(bl / Lk);
  Vt[((size_t)(b * 4 + kvh) * 64 + hd) * (size_t)Lk + l] = V[idx];
}

__global__ __launch_bounds__(256) void rmsnorm_kernel(
    const float* __restrict__ X, const float* __restrict__ g,
    bf16_t* __restrict__ Y, float* __restrict__ Yf, int D) {
  int row = blockIdx.x;
  int tid = threadIdx.x;
  const float* x = X + (size_t)row * D;
  float ss = 0.f;
  for (int i = tid; i < D; i += 256) { float v = x[i]; ss += v * v; }
#pragma unroll
  for (int sh = 16; sh >= 1; sh >>= 1) ss += __shfl_xor(ss, sh, 32);
  __shared__ float red[8];
  if ((tid & 31) == 0) red[tid >> 5] = ss;
  __syncthreads();
  float tot = 0.f;
#pragma unroll
  for (int i = 0; i < 8; ++i) tot += red[i];
  float scale = rsqrtf(tot / D + 1e-6f);
  for (int i = tid; i < D; i += 256) {
    float v = x[i] * scale * g[i];
    Y[(size_t)row * D + i] = (bf16_t)v;
    if (Yf) Yf[(size_t)row * D + i] = v;
  }
}

// router: one block (128 thr) per token. logits = xn @ Wr[1024,8]; softmax;
// top-2 (ties -> lowest index like lax.top_k); combine weights; aux sums.
__global__ __launch_bounds__(128) void router_kernel(
    const float* __restrict__ Xn, const float* __restrict__ Wr,
    float* __restrict__ cw, float* __restrict__ sums, int D) {
  int t = blockIdx.x;
  int tid = threadIdx.x;
  float acc[8];
#pragma unroll
  for (int e = 0; e < 8; ++e) acc[e] = 0.f;
  for (int i = tid; i < D; i += 128) {
    float xv = Xn[(size_t)t * D + i];
#pragma unroll
    for (int e = 0; e < 8; ++e) acc[e] += xv * Wr[(size_t)i * 8 + e];
  }
#pragma unroll
  for (int sh = 16; sh >= 1; sh >>= 1)
#pragma unroll
    for (int e = 0; e < 8; ++e) acc[e] += __shfl_xor(acc[e], sh, 32);
  __shared__ float red[4][8];
  if ((tid & 31) == 0)
#pragma unroll
    for (int e = 0; e < 8; ++e) red[tid >> 5][e] = acc[e];
  __syncthreads();
  if (tid == 0) {
    float lg[8];
#pragma unroll
    for (int e = 0; e < 8; ++e)
      lg[e] = red[0][e] + red[1][e] + red[2][e] + red[3][e];
    float mx = lg[0];
#pragma unroll
    for (int e = 1; e < 8; ++e) mx = fmaxf(mx, lg[e]);
    float p[8], se = 0.f;
#pragma unroll
    for (int e = 0; e < 8; ++e) { p[e] = __expf(lg[e] - mx); se += p[e]; }
#pragma unroll
    for (int e = 0; e < 8; ++e) p[e] /= se;
    int i1 = 0;
#pragma unroll
    for (int e = 1; e < 8; ++e) if (p[e] > p[i1]) i1 = e;
    int i2 = (i1 == 0) ? 1 : 0;
#pragma unroll
    for (int e = 0; e < 8; ++e)
      if (e != i1 && p[e] > p[i2]) i2 = e;
    float s2 = p[i1] + p[i2];
#pragma unroll
    for (int e = 0; e < 8; ++e) cw[(size_t)t * 8 + e] = 0.f;
    cw[(size_t)t * 8 + i1] = p[i1] / s2;
    cw[(size_t)t * 8 + i2] = p[i2] / s2;
    atomicAdd(&sums[i1], 1.f);
    atomicAdd(&sums[i2], 1.f);
#pragma unroll
    for (int e = 0; e < 8; ++e) atomicAdd(&sums[8 + e], p[e]);
  }
}

__global__ void aux_final_kernel(const float* __restrict__ sums,
                                 float* __restrict__ out, int T) {
  float a = 0.f;
#pragma unroll
  for (int e = 0; e < 8; ++e) {
    float f = sums[e] / ((float)T * 2.f);
    float pm = sums[8 + e] / (float)T;
    a += f * pm;
  }
  out[0] = 8.f * a;
}

// ---------------------------------------------------------------- host glue
extern "C" void kernel_launch(void* const* d_in, const int* in_sizes, int n_in,
                              void* d_out, int out_size, void* d_ws,
                              size_t ws_size, hipStream_t stream) {
  const int Dm = 1024, Ff = 2048, T = 2048, Sq = 1024, Lk = 1024;
  const float* x = (const float*)d_in[0];
  const float* enc = (const float*)d_in[1];
  const float* g1 = (const float*)d_in[2];
  const float* g2 = (const float*)d_in[3];
  const float* g3 = (const float*)d_in[4];
  const float* sWq = (const float*)d_in[5];
  const float* sWk = (const float*)d_in[6];
  const float* sWv = (const float*)d_in[7];
  const float* sWo = (const float*)d_in[8];
  const float* cWq = (const float*)d_in[9];
  const float* cWk = (const float*)d_in[10];
  const float* cWv = (const float*)d_in[11];
  const float* cWo = (const float*)d_in[12];
  const float* Wr = (const float*)d_in[13];
  const float* Wgate = (const float*)d_in[14];
  const float* Wup = (const float*)d_in[15];
  const float* Wdown = (const float*)d_in[16];
  float* outp = (float*)d_out;

  char* base = (char*)d_ws;
  size_t off = 0;
  auto walloc = [&](size_t bytes) -> void* {
    void* p = base + off;
    off = (off + bytes + 255) & ~(size_t)255;
    return p;
  };
  const size_t D2 = (size_t)Dm * Dm;       // 1M
  const size_t DF = (size_t)Dm * Ff;       // 2M
  bf16_t* wq_s = (bf16_t*)walloc(D2 * 2);
  bf16_t* wk_s = (bf16_t*)walloc((size_t)Dm * 256 * 2);
  bf16_t* wv_s = (bf16_t*)walloc((size_t)Dm * 256 * 2);
  bf16_t* wo_s = (bf16_t*)walloc(D2 * 2);
  bf16_t* wq_c = (bf16_t*)walloc(D2 * 2);
  bf16_t* wk_c = (bf16_t*)walloc((size_t)Dm * 256 * 2);
  bf16_t* wv_c = (bf16_t*)walloc((size_t)Dm * 256 * 2);
  bf16_t* wo_c = (bf16_t*)walloc(D2 * 2);
  bf16_t* wgate_t = (bf16_t*)walloc(8 * DF * 2);
  bf16_t* wup_t = (bf16_t*)walloc(8 * DF * 2);
  bf16_t* wdown_t = (bf16_t*)walloc(8 * DF * 2);
  bf16_t* enc_bf = (bf16_t*)walloc((size_t)T * Dm * 2);
  bf16_t* n1 = (bf16_t*)walloc((size_t)T * Dm * 2);
  bf16_t* n2 = (bf16_t*)walloc((size_t)T * Dm * 2);
  bf16_t* n3 = (bf16_t*)walloc((size_t)T * Dm * 2);
  float* n3f = (float*)walloc((size_t)T * Dm * 4);
  bf16_t* qb = (bf16_t*)walloc((size_t)T * Dm * 2);
  bf16_t* kb1 = (bf16_t*)walloc((size_t)T * 256 * 2);
  bf16_t* vb = (bf16_t*)walloc((size_t)T * 256 * 2);
  bf16_t* vt1 = (bf16_t*)walloc((size_t)T * 256 * 2);
  bf16_t* ab = (bf16_t*)walloc((size_t)T * Dm * 2);
  bf16_t* kb2 = (bf16_t*)walloc((size_t)T * 256 * 2);
  bf16_t* vt2 = (bf16_t*)walloc((size_t)T * 256 * 2);
  float* hf = (float*)walloc((size_t)T * Dm * 4);
  float* h2f = (float*)walloc((size_t)T * Dm * 4);
  float* cw = (float*)walloc((size_t)T * 8 * 4);
  float* sums = (float*)walloc(16 * 4);
  float* gbuf = (float*)walloc((size_t)T * Ff * 4);
  bf16_t* hbb = (bf16_t*)walloc((size_t)T * Ff * 2);

  auto ewg = [](long n) { return dim3((unsigned)((n + 255) / 256)); };

  // 1) weight conversion (fp32 [K,N] -> bf16 [N,K])
  convert_w_t_kernel<<<ewg((long)D2), 256, 0, stream>>>(sWq, wq_s, Dm, Dm, (long)D2);
  convert_w_t_kernel<<<ewg((long)Dm * 256), 256, 0, stream>>>(sWk, wk_s, Dm, 256, (long)Dm * 256);
  convert_w_t_kernel<<<ewg((long)Dm * 256), 256, 0, stream>>>(sWv, wv_s, Dm, 256, (long)Dm * 256);
  convert_w_t_kernel<<<ewg((long)D2), 256, 0, stream>>>(sWo, wo_s, Dm, Dm, (long)D2);
  convert_w_t_kernel<<<ewg((long)D2), 256, 0, stream>>>(cWq, wq_c, Dm, Dm, (long)D2);
  convert_w_t_kernel<<<ewg((long)Dm * 256), 256, 0, stream>>>(cWk, wk_c, Dm, 256, (long)Dm * 256);
  convert_w_t_kernel<<<ewg((long)Dm * 256), 256, 0, stream>>>(cWv, wv_c, Dm, 256, (long)Dm * 256);
  convert_w_t_kernel<<<ewg((long)D2), 256, 0, stream>>>(cWo, wo_c, Dm, Dm, (long)D2);
  convert_w_t_kernel<<<ewg((long)8 * DF), 256, 0, stream>>>(Wgate, wgate_t, Dm, Ff, (long)8 * DF);
  convert_w_t_kernel<<<ewg((long)8 * DF), 256, 0, stream>>>(Wup, wup_t, Dm, Ff, (long)8 * DF);
  convert_w_t_kernel<<<ewg((long)8 * DF), 256, 0, stream>>>(Wdown, wdown_t, Ff, Dm, (long)8 * DF);
  cast_bf16_kernel<<<ewg((long)T * Dm), 256, 0, stream>>>(enc, enc_bf, (long)T * Dm);

  dim3 gQ(Dm / 128, T / 128);     // N=1024
  dim3 gKV(256 / 128, T / 128);   // N=256
  dim3 gFF(Ff / 128, T / 128);    // N=2048
  dim3 gD(Dm / 128, T / 128);     // N=1024 (down)
  dim3 blk(256);

  // 2) self-attention
  rmsnorm_kernel<<<T, 256, 0, stream>>>(x, g1, n1, nullptr, Dm);
  gemm_wmma_kernel<false, false, false, true, false><<<gQ, blk, 0, stream>>>(
      n1, wq_s, qb, nullptr, nullptr, nullptr, 0, Dm, Dm, Dm, Dm);
  gemm_wmma_kernel<false, false, false, true, false><<<gKV, blk, 0, stream>>>(
      n1, wk_s, kb1, nullptr, nullptr, nullptr, 0, Dm, Dm, Dm, 256);
  gemm_wmma_kernel<false, false, false, true, false><<<gKV, blk, 0, stream>>>(
      n1, wv_s, vb, nullptr, nullptr, nullptr, 0, Dm, Dm, Dm, 256);
  transpose_v_kernel<<<ewg((long)T * 256), 256, 0, stream>>>(vb, vt1, Sq, (long)T * 256);
  flash_attn_kernel<<<dim3(Sq / 64, 16, 2), dim3(128), 0, stream>>>(
      qb, kb1, vt1, ab, Sq, Sq, 1);
  gemm_wmma_kernel<false, true, false, false, false><<<gQ, blk, 0, stream>>>(
      ab, wo_s, hf, x, nullptr, nullptr, 0, Dm, Dm, Dm, Dm);

  // 3) cross-attention
  rmsnorm_kernel<<<T, 256, 0, stream>>>(hf, g2, n2, nullptr, Dm);
  gemm_wmma_kernel<false, false, false, true, false><<<gQ, blk, 0, stream>>>(
      n2, wq_c, qb, nullptr, nullptr, nullptr, 0, Dm, Dm, Dm, Dm);
  gemm_wmma_kernel<false, false, false, true, false><<<gKV, blk, 0, stream>>>(
      enc_bf, wk_c, kb2, nullptr, nullptr, nullptr, 0, Dm, Dm, Dm, 256);
  gemm_wmma_kernel<false, false, false, true, false><<<gKV, blk, 0, stream>>>(
      enc_bf, wv_c, vb, nullptr, nullptr, nullptr, 0, Dm, Dm, Dm, 256);
  transpose_v_kernel<<<ewg((long)T * 256), 256, 0, stream>>>(vb, vt2, Lk, (long)T * 256);
  flash_attn_kernel<<<dim3(Sq / 64, 16, 2), dim3(128), 0, stream>>>(
      qb, kb2, vt2, ab, Sq, Lk, 0);
  gemm_wmma_kernel<false, true, false, false, false><<<gQ, blk, 0, stream>>>(
      ab, wo_c, h2f, hf, nullptr, nullptr, 0, Dm, Dm, Dm, Dm);

  // 4) MoE
  rmsnorm_kernel<<<T, 256, 0, stream>>>(h2f, g3, n3, n3f, Dm);
  hipMemsetAsync(sums, 0, 16 * sizeof(float), stream);
  router_kernel<<<T, 128, 0, stream>>>(n3f, Wr, cw, sums, Dm);
  for (int e = 0; e < 8; ++e) {
    // gate -> fp32 gbuf
    gemm_wmma_kernel<false, false, false, false, false><<<gFF, blk, 0, stream>>>(
        n3, wgate_t + (size_t)e * DF, gbuf, nullptr, nullptr, nullptr, 0, Dm,
        Dm, Dm, Ff);
    // up, fused SwiGLU epilogue: hbb = bf16(silu(gbuf) * up)
    gemm_wmma_kernel<false, false, false, true, true><<<gFF, blk, 0, stream>>>(
        n3, wup_t + (size_t)e * DF, hbb, nullptr, gbuf, nullptr, 0, Dm, Dm, Dm,
        Ff);
    // down, scaled by combine weight; e==0 seeds with residual h2
    if (e == 0) {
      gemm_wmma_kernel<false, true, true, false, false><<<gD, blk, 0, stream>>>(
          hbb, wdown_t + (size_t)e * DF, outp, h2f, nullptr, cw + e, 8, Ff, Ff,
          Ff, Dm);
    } else {
      gemm_wmma_kernel<true, false, true, false, false><<<gD, blk, 0, stream>>>(
          hbb, wdown_t + (size_t)e * DF, outp, nullptr, nullptr, cw + e, 8, Ff,
          Ff, Ff, Dm);
    }
  }
  aux_final_kernel<<<1, 1, 0, stream>>>(sums, outp + (size_t)T * Dm, T);
}
